// TemporalGraphModel_6966436954805
// MI455X (gfx1250) — compile-verified, matrix operand order checked by reading, not run on hardware
//
#include <hip/hip_runtime.h>
#include <cstdint>
#include <cstddef>

// ---- problem dims ----
#define BB    8
#define SS    12
#define NN    4096
#define FF    4
#define HH    64
#define CATF  68            // F + H
#define COLPB 72            // padded columns per batch
#define NCOL  576           // COLPB * BB  (36 tiles of 16)
#define NH    262144        // N*H
#define K4    4096
#define LDA   72            // padded LDS row stride (elems) = 144B -> spreads banks, 16B aligned

typedef __bf16 bf16_t;
typedef __attribute__((ext_vector_type(16))) __bf16       v16bf;
typedef __attribute__((ext_vector_type(8)))  float        v8f;
typedef __attribute__((ext_vector_type(4)))  unsigned int u32x4;

union Frag { v16bf v; u32x4 u[2]; };
union Acc  { v8f   v; float f[8]; };

__device__ __forceinline__ float sigmoidf_(float x) { return 1.0f / (1.0f + __expf(-x)); }

// async global->LDS copy, 16 bytes per lane (tracked by ASYNCcnt).
// VDST = VGPR holding LDS byte offset; VADDR = 64-bit global address (GV mode).
__device__ __forceinline__ void async_ld_b128(unsigned lds_off, const void* gptr) {
    asm volatile("global_load_async_to_lds_b128 %0, %1, off"
                 :: "v"(lds_off), "v"(gptr) : "memory");
}
__device__ __forceinline__ unsigned lds_off_of(const void* p) {
    return (unsigned)(size_t)p;   // generic LDS pointer: addr[31:0] is the LDS offset
}

// ---------- one-time prep ----------
__global__ void k_f32_to_bf16(const float* __restrict__ src, bf16_t* __restrict__ dst, int n) {
    int i = blockIdx.x * blockDim.x + threadIdx.x;
    if (i < n) dst[i] = (bf16_t)src[i];
}

__global__ void k_init_h(const float* __restrict__ se, float* __restrict__ h) {
    int i = blockIdx.x * blockDim.x + threadIdx.x;        // [0, BB*NH)
    if (i < BB * NH) h[i] = se[i & (NH - 1)];
}

__global__ void k_zero(float* __restrict__ p, int n) {
    int i = blockIdx.x * blockDim.x + threadIdx.x;
    if (i < n) p[i] = 0.0f;
}

__global__ void k_copy(const float* __restrict__ s, float* __restrict__ d, int n) {
    int i = blockIdx.x * blockDim.x + threadIdx.x;
    if (i < n) d[i] = s[i];
}

// ---------- encoder: pack cat^T [NCOL][K4] bf16 ----------
__global__ void k_pack_cat(const float* __restrict__ X, const float* __restrict__ h,
                           bf16_t* __restrict__ catT, int s) {
    int i = blockIdx.x * blockDim.x + threadIdx.x;        // [0, NCOL*NN)
    if (i >= NCOL * NN) return;
    int n   = i & (NN - 1);
    int col = i >> 12;
    int b   = col / COLPB;
    int f   = col - b * COLPB;
    float v = 0.0f;
    if (f < FF)        v = X[(((size_t)b * SS + s) * NN + n) * FF + f];
    else if (f < CATF) v = h[(size_t)b * NH + (size_t)n * HH + (f - FF)];
    catT[(size_t)col * K4 + n] = (bf16_t)v;
}

// ---------- encoder GEMM: a[4096][576] = adj_bf16 @ cat ----------
// Block (256 thr, 8 waves) computes a 128(M) x 32(N) macro-tile.
// Per K-chunk (64): stage A 128x64 + B 32x64 into LDS via async global->LDS
// (double-buffered, ASYNCcnt-pipelined), then each wave does a 16x32 tile:
// 2 fp32 accumulators, ds_load_b128 fragment reads, v_wmma_f32_16x16x32_bf16.
__global__ __launch_bounds__(256) void k_gemm_diffuse(const bf16_t* __restrict__ adjb,
                                                      const bf16_t* __restrict__ catT,
                                                      float* __restrict__ aout) {
    __shared__ __align__(16) bf16_t sA[2][128 * LDA];   // 2 x 18KB
    __shared__ __align__(16) bf16_t sB[2][32 * LDA];    // 2 x 4.5KB

    const int t      = threadIdx.x;
    const int lane   = t & 31;
    const int w      = t >> 5;
    const int laneLo = lane & 15;
    const int kHi    = (lane >> 4) << 3;                // 0 or 8
    const int blkM   = blockIdx.x;                      // 32
    const int blkN   = blockIdx.y;                      // 18

    const bf16_t* gA = adjb + (size_t)blkM * 128 * K4;
    const bf16_t* gB = catT + (size_t)blkN * 32 * K4;

    // staging map: 16B segments; A: 4 per thread, B: 1 per thread
    const int aRow0 = (t << 2) >> 3;                    // rows for this thread's 4 A segs
    const int bRow  = t >> 3;
    const int bSc   = t & 7;

    auto stage = [&](int bi, int kc) {
        unsigned baseA = lds_off_of(&sA[bi][0]);
        #pragma unroll
        for (int q = 0; q < 4; ++q) {
            int sidx = (t << 2) + q;
            int row  = sidx >> 3;
            int sc   = sidx & 7;
            async_ld_b128(baseA + (unsigned)(row * LDA + sc * 8) * 2,
                          gA + (size_t)row * K4 + kc + sc * 8);
        }
        unsigned baseB = lds_off_of(&sB[bi][0]);
        async_ld_b128(baseB + (unsigned)(bRow * LDA + bSc * 8) * 2,
                      gB + (size_t)bRow * K4 + kc + bSc * 8);
    };

    v8f zv = {0.f, 0.f, 0.f, 0.f, 0.f, 0.f, 0.f, 0.f};
    Acc acc0, acc1;
    acc0.v = zv; acc1.v = zv;

    const int mRow = w * 16 + laneLo;                   // A fragment row within macro-tile

    stage(0, 0);
    int bi = 0;
    for (int kc = 0; kc < K4; kc += 64, bi ^= 1) {
        if (kc + 64 < K4) {
            stage(bi ^ 1, kc + 64);
            asm volatile("s_wait_asynccnt 0x5" ::: "memory");  // current buffer's 5 done
        } else {
            asm volatile("s_wait_asynccnt 0x0" ::: "memory");
        }
        __syncthreads();

        #pragma unroll
        for (int ks = 0; ks < 64; ks += 32) {
            Frag a, b0, b1;
            const bf16_t* ap  = &sA[bi][mRow * LDA + ks + kHi];
            const bf16_t* bp0 = &sB[bi][laneLo * LDA + ks + kHi];
            const bf16_t* bp1 = &sB[bi][(16 + laneLo) * LDA + ks + kHi];
            a.u[0]  = *(const u32x4*)(ap);
            a.u[1]  = *(const u32x4*)(ap + 16);
            b0.u[0] = *(const u32x4*)(bp0);
            b0.u[1] = *(const u32x4*)(bp0 + 16);
            b1.u[0] = *(const u32x4*)(bp1);
            b1.u[1] = *(const u32x4*)(bp1 + 16);
            acc0.v = __builtin_amdgcn_wmma_f32_16x16x32_bf16(false, a.v, false, b0.v,
                                                             (short)0, acc0.v, false, false);
            acc1.v = __builtin_amdgcn_wmma_f32_16x16x32_bf16(false, a.v, false, b1.v,
                                                             (short)0, acc1.v, false, false);
        }
        __syncthreads();                                // cur buffer fully read before reuse
    }

    // 32-bit C/D layout: VGPR r -> M = r (lanes 0-15) / r+8 (lanes 16-31), N = lane%16
    int rBase = blkM * 128 + w * 16 + ((lane >> 4) << 3);
    int c0    = blkN * 32 + laneLo;
    #pragma unroll
    for (int r = 0; r < 8; ++r) {
        aout[(size_t)(rBase + r) * NCOL + c0]      = acc0.f[r];
        aout[(size_t)(rBase + r) * NCOL + c0 + 16] = acc1.f[r];
    }
    (void)aRow0;
}

// ---------- encoder: cell update with the reference's flat-split gate quirk ----------
__global__ __launch_bounds__(256) void k_cell(const float* __restrict__ aout,
                       const float* __restrict__ W1, const float* __restrict__ b1,
                       const float* __restrict__ W2, const float* __restrict__ b2,
                       const float* __restrict__ se, float* __restrict__ hNext) {
    int i = blockIdx.x * blockDim.x + threadIdx.x;        // [0, BB*NH)
    if (i >= BB * NH) return;
    int b = i >> 18;
    int t = i & (NH - 1);
    const float* arow_b = aout + b * COLPB;

    // conv = a[b,n,:] @ W2[:,h] + b2[h]
    int n = t >> 6, hh = t & 63;
    const float* ar = arow_b + (size_t)n * NCOL;
    float accC = b2[hh];
    for (int f = 0; f < CATF; ++f) accC += ar[f] * W2[f * HH + hh];

    // gates: flat split of [B, N*3H] -> gate g at flat index u = g*NH + t
    float g[3];
    #pragma unroll
    for (int gi = 0; gi < 3; ++gi) {
        int u  = gi * NH + t;
        int np = u / 192;
        int j  = u - np * 192;
        const float* ag = arow_b + (size_t)np * NCOL;
        float acc = b1[j];
        for (int f = 0; f < CATF; ++f) acc += ag[f] * W1[f * 192 + j];
        g[gi] = sigmoidf_(acc);
    }
    float c = g[0] * se[t] + g[1] * tanhf(accC);          // c0 == struc_emb forever
    hNext[i] = g[2] * tanhf(c);
}

// ---------- decoder: constant input contribution ----------
__global__ void k_xgates(const float* __restrict__ z, const float* __restrict__ Wih,
                         const float* __restrict__ bih, const float* __restrict__ bhh,
                         float* __restrict__ xg) {
    int i = blockIdx.x * blockDim.x + threadIdx.x;        // [0, 32768*256)
    if (i >= 32768 * 256) return;
    int r = i >> 8, j = i & 255;
    float acc = bih[j] + bhh[j];
    const float* zr = z + (size_t)r * HH;
    const float* w  = Wih + (size_t)j * HH;
    for (int k = 0; k < HH; ++k) acc += zr[k] * w[k];
    xg[i] = acc;
}

// ---------- decoder: one LSTM step + fused MLP head, one block per row ----------
__global__ __launch_bounds__(64) void k_dec_step(const float* __restrict__ xg,
        const float* __restrict__ Whh, const float* __restrict__ D1,
        const float* __restrict__ bd1, const float* __restrict__ D2,
        const float* __restrict__ bd2, float* __restrict__ hdec,
        float* __restrict__ cdec, float* __restrict__ recon, int s) {
    __shared__ float sh[HH], shn[HH], sm[32];
    int r   = blockIdx.x;                                  // [0, 32768)
    int tid = threadIdx.x;                                 // [0, 64)
    sh[tid] = hdec[(size_t)r * HH + tid];
    __syncthreads();

    float gate[4];                                         // torch order: i, f, g, o
    #pragma unroll
    for (int gi = 0; gi < 4; ++gi) {
        int j = gi * HH + tid;
        float acc = xg[(size_t)r * 256 + j];
        const float* w = Whh + (size_t)j * HH;
        for (int k = 0; k < HH; ++k) acc += sh[k] * w[k];
        gate[gi] = acc;
    }
    float cOld = cdec[(size_t)r * HH + tid];
    float c    = sigmoidf_(gate[1]) * cOld + sigmoidf_(gate[0]) * tanhf(gate[2]);
    float hNew = sigmoidf_(gate[3]) * tanhf(c);
    cdec[(size_t)r * HH + tid] = c;
    hdec[(size_t)r * HH + tid] = hNew;
    shn[tid] = hNew;
    __syncthreads();

    if (tid < 32) {
        float m = bd1[tid];
        for (int k = 0; k < HH; ++k) m += shn[k] * D1[k * 32 + tid];
        sm[tid] = m > 0.f ? m : 0.f;
    }
    __syncthreads();
    if (tid == 0) {
        float o = bd2[0];
        for (int k = 0; k < 32; ++k) o += sm[k] * D2[k];
        recon[(size_t)r * SS + s] = o;                     // (B*N,S)->(B,S,N) flat reshape
    }
}

extern "C" void kernel_launch(void* const* d_in, const int* in_sizes, int n_in,
                              void* d_out, int out_size, void* d_ws, size_t ws_size,
                              hipStream_t stream) {
    const float* X   = (const float*)d_in[0];
    const float* adj = (const float*)d_in[1];
    const float* se  = (const float*)d_in[2];
    const float* W1  = (const float*)d_in[3];
    const float* b1  = (const float*)d_in[4];
    const float* W2  = (const float*)d_in[5];
    const float* b2  = (const float*)d_in[6];
    const float* Wih = (const float*)d_in[7];
    const float* Whh = (const float*)d_in[8];
    const float* bih = (const float*)d_in[9];
    const float* bhh = (const float*)d_in[10];
    const float* D1  = (const float*)d_in[11];
    const float* bd1 = (const float*)d_in[12];
    const float* D2  = (const float*)d_in[13];
    const float* bd2 = (const float*)d_in[14];
    float* out = (float*)d_out;

    char* ws = (char*)d_ws;
    size_t off = 0;
    auto alloc = [&](size_t bytes) -> void* {
        void* p = ws + off;
        off += (bytes + 255) & ~(size_t)255;
        return p;
    };
    bf16_t* adjb  = (bf16_t*)alloc((size_t)NN * NN * 2);
    bf16_t* catT  = (bf16_t*)alloc((size_t)NCOL * K4 * 2);
    float*  aoutw = (float*)alloc((size_t)NN * NCOL * 4);
    float*  hA    = (float*)alloc((size_t)BB * NH * 4);
    float*  hB    = (float*)alloc((size_t)BB * NH * 4);
    float*  hdec  = (float*)alloc((size_t)32768 * HH * 4);
    float*  cdec  = (float*)alloc((size_t)32768 * HH * 4);
    float*  xg    = (float*)alloc((size_t)32768 * 256 * 4);

    k_f32_to_bf16<<<(NN * NN + 255) / 256, 256, 0, stream>>>(adj, adjb, NN * NN);
    k_init_h<<<(BB * NH + 255) / 256, 256, 0, stream>>>(se, hA);

    float* hCur = hA; float* hNxt = hB;
    for (int s = 0; s < SS; ++s) {
        k_pack_cat<<<(NCOL * NN + 255) / 256, 256, 0, stream>>>(X, hCur, catT, s);
        k_gemm_diffuse<<<dim3(32, 18), 256, 0, stream>>>(adjb, catT, aoutw);
        k_cell<<<(BB * NH + 255) / 256, 256, 0, stream>>>(aoutw, W1, b1, W2, b2, se, hNxt);
        float* tmp = hCur; hCur = hNxt; hNxt = tmp;
    }

    // z output = h_last (layout already [B][N*H] flat)
    k_copy<<<(BB * NH + 255) / 256, 256, 0, stream>>>(hCur, out, BB * NH);

    // decoder
    k_xgates<<<(32768 * 256 + 255) / 256, 256, 0, stream>>>(hCur, Wih, bih, bhh, xg);
    k_zero<<<(32768 * HH + 255) / 256, 256, 0, stream>>>(hdec, 32768 * HH);
    k_zero<<<(32768 * HH + 255) / 256, 256, 0, stream>>>(cdec, 32768 * HH);
    for (int s = 0; s < SS; ++s) {
        k_dec_step<<<32768, 64, 0, stream>>>(xg, Whh, D1, bd1, D2, bd2,
                                             hdec, cdec, out + BB * NH, s);
    }
    (void)in_sizes; (void)n_in; (void)out_size; (void)ws_size;
}
